// StableFourierConvSSM_13091060318292
// MI455X (gfx1250) — compile-verified, matrix operand order checked by reading, not run on hardware
//
#include <hip/hip_runtime.h>
#include <hip/hip_bf16.h>

// ---------------------------------------------------------------------------
// StableFourierConvSSM for MI455X (gfx1250, wave32, WMMA + async-to-LDS)
// B=16, H=W=128, C=64, C2=128, K=7, T=16, N = B*H*W = 262144
//
// dft_stab_kernel: 49-tap direct DFT of the padded 7x7 complex kernels + A
//   stabilization -> 16 MB of f32 tables in workspace.
// fused_gates_scan_kernel: per 16-row tile,
//   - async-copies the f32 x tile global->LDS (global_load_async_to_lds_b128,
//     ASYNCcnt), so x is read from HBM exactly once,
//   - stages the three weights transposed as f16 in LDS (padded rows, no
//     16-way bank conflicts on fragment loads),
//   - 3 gate GEMMs as 96 v_wmma_f32_16x16x32_f16 per wave,
//   - fast sigmoid/softplus (v_rcp/v_exp/v_log), gates staged in LDS,
//   - complex input transform + fully unrolled T=16 scan, one HBM write.
// HBM traffic ~ 134 MB (x) + 134 MB (out) + 16 MB tables -> memory-bound
// optimum at 23.3 TB/s.
// ---------------------------------------------------------------------------

#define Bsz   16
#define Hdim  128
#define Wdim  128
#define Cdim  64
#define C2    128
#define Kk    7
#define Tlen  16
#define HW    (Hdim * Wdim)
#define NROWS (Bsz * HW)

// padded LDS row strides (in halfs): 136 halfs = 272 B, 16B-aligned, lanes
// land 4 banks apart instead of 0 -> kills the 16-way conflicts.
#define WT_STRIDE 136
#define SX_STRIDE 136

// LDS layout (byte offsets into dynamic smem)
#define WT_OFF    0                       // 3 * 128 * 136 halfs = 104448 B
#define SX_OFF    104448                  // 4 * 16 * 136 halfs  =  17408 B
#define SG_OFF    121856                  // 4 * 3 * 16 * 128 halfs = 49152 B
#define SXF32_OFF 171008                  // 4 * 16 * 128 f32    =  32768 B
#define SMEM_BYTES 203776

typedef __attribute__((ext_vector_type(16))) _Float16 v16h;
typedef __attribute__((ext_vector_type(8)))  _Float16 v8h;
typedef __attribute__((ext_vector_type(8)))  float    v8f;

union Frag16 { v16h v; v8h h[2]; };

__device__ __forceinline__ float fast_sigmoid(float x) {
    // v_exp_f32 + v_rcp_f32 instead of the IEEE div chain
    return __builtin_amdgcn_rcpf(1.0f + __expf(-x));
}
__device__ __forceinline__ float fast_softplus(float x) {
    return (x > 15.0f) ? x : __logf(1.0f + __expf(x));
}

// ---------------------------------------------------------------------------
// Kernel 1: direct DFT of padded kernels + A stabilization.
// grid = H*W blocks, 64 threads (one per channel c).
// F[u,v] = sum_{p,q} kern[c,p,q] * exp(-2*pi*i*(u*(p+60) + v*(q+60))/128)
// ---------------------------------------------------------------------------
__global__ void dft_stab_kernel(const float* __restrict__ A_real,
                                const float* __restrict__ A_imag,
                                const float* __restrict__ B_real,
                                const float* __restrict__ B_imag,
                                float* __restrict__ wsAr,
                                float* __restrict__ wsAi,
                                float* __restrict__ wsBr,
                                float* __restrict__ wsBi) {
    const int hw = blockIdx.x;          // 0..16383
    const int u  = hw >> 7;
    const int v  = hw & 127;
    const int c  = threadIdx.x;         // 0..63
    const float twoPiOver128 = 6.283185307179586f / 128.0f;

    float eur[Kk], eui[Kk], evr[Kk], evi[Kk];
#pragma unroll
    for (int p = 0; p < Kk; ++p) {
        int tu = (u * (p + 60)) & 127;  // exact phase reduction mod 128
        int tv = (v * (p + 60)) & 127;
        float su, cu, sv, cv;
        __sincosf(-twoPiOver128 * (float)tu, &su, &cu);
        __sincosf(-twoPiOver128 * (float)tv, &sv, &cv);
        eur[p] = cu; eui[p] = su;
        evr[p] = cv; evi[p] = sv;
    }

    const float* Arp = A_real + c * (Kk * Kk);
    const float* Aip = A_imag + c * (Kk * Kk);
    const float* Brp = B_real + c * (Kk * Kk);
    const float* Bip = B_imag + c * (Kk * Kk);

    float ar = 0.f, ai = 0.f, br = 0.f, bi = 0.f;
#pragma unroll
    for (int p = 0; p < Kk; ++p) {
#pragma unroll
        for (int q = 0; q < Kk; ++q) {
            float tr = eur[p] * evr[q] - eui[p] * evi[q];
            float ti = eur[p] * evi[q] + eui[p] * evr[q];
            float kr = Arp[p * Kk + q], ki = Aip[p * Kk + q];
            ar += kr * tr - ki * ti;
            ai += kr * ti + ki * tr;
            kr = Brp[p * Kk + q]; ki = Bip[p * Kk + q];
            br += kr * tr - ki * ti;
            bi += kr * ti + ki * tr;
        }
    }

    // stabilize A: scale = 0.95*sigmoid(|A|)/(|A|+1e-8)
    float mag   = sqrtf(ar * ar + ai * ai + 1e-8f);
    float scale = 0.95f * fast_sigmoid(mag) * __builtin_amdgcn_rcpf(mag + 1e-8f);

    const int idx = hw * Cdim + c;
    wsAr[idx] = ar * scale;
    wsAi[idx] = ai * scale;
    wsBr[idx] = br;
    wsBi[idx] = bi;
}

// ---------------------------------------------------------------------------
// Kernel 2: fused gate GEMMs + activations + scan.
// Block = 128 threads (4 wave32), each wave owns a 16-row x 128-col tile.
// ---------------------------------------------------------------------------
__global__ void fused_gates_scan_kernel(const float* __restrict__ x,
                                        const float* __restrict__ Wf,
                                        const float* __restrict__ bf_p,
                                        const float* __restrict__ Wi,
                                        const float* __restrict__ bi_p,
                                        const float* __restrict__ Wd,
                                        const float* __restrict__ bd_p,
                                        const float* __restrict__ wsAr,
                                        const float* __restrict__ wsAi,
                                        const float* __restrict__ wsBr,
                                        const float* __restrict__ wsBi,
                                        float* __restrict__ out) {
    extern __shared__ char smem[];
    _Float16* sWt   = (_Float16*)(smem + WT_OFF);
    _Float16* sX    = (_Float16*)(smem + SX_OFF);
    _Float16* sG    = (_Float16*)(smem + SG_OFF);
    float*    sXf32 = (float*)   (smem + SXF32_OFF);

    const int tid  = threadIdx.x;
    const int wave = tid >> 5;
    const int lane = tid & 31;
    const int nl   = lane & 15;         // N index within 16x16 tile
    const int hi   = lane >> 4;         // lane-half selector
    const int row0 = (blockIdx.x * 4 + wave) * 16;

    const float* Wg[3] = { Wf, Wi, Wd };
    const float* bg[3] = { bf_p, bi_p, bd_p };

    // L2 prefetch hint for the (L2-resident) weights
    __builtin_prefetch(Wf + tid * 4, 0, 1);
    __builtin_prefetch(Wi + tid * 4, 0, 1);
    __builtin_prefetch(Wd + tid * 4, 0, 1);

    // ---- Async-copy this wave's 16x128 f32 x tile global -> LDS.
    // Per-lane 16B chunks, tracked by ASYNCcnt. x is read from HBM once.
    {
        float* dstBase = sXf32 + wave * (16 * C2);
        const float* srcBase = x + (size_t)row0 * C2;
#pragma unroll
        for (int i = 0; i < 16; ++i) {
            const int idx16 = i * 32 + lane;              // 16B chunk index
            unsigned ldsAddr = (unsigned)(uintptr_t)(dstBase + idx16 * 4);
            unsigned long long gAddr =
                (unsigned long long)(uintptr_t)(srcBase + idx16 * 4);
            asm volatile("global_load_async_to_lds_b128 %0, %1, off"
                         :: "v"(ldsAddr), "v"(gAddr) : "memory");
        }
    }

    // ---- Stage transposed weights (f16) cooperatively: Wt[g][n][k] = W[g][k][n]
    for (int g = 0; g < 3; ++g) {
        const float* Wsrc = Wg[g];
        for (int idx = tid; idx < C2 * C2; idx += 128) {
            int k = idx >> 7;
            int n = idx & 127;
            sWt[g * (C2 * WT_STRIDE) + n * WT_STRIDE + k] = (_Float16)Wsrc[idx];
        }
    }

    // Wait for the async x tile, then build the f16 copy used by WMMA.
    asm volatile("s_wait_asynccnt 0x0" ::: "memory");
    {
        const float* srcT = sXf32 + wave * (16 * C2);
        _Float16*    dstT = sX    + wave * (16 * SX_STRIDE);
        for (int i = lane; i < 16 * C2; i += 32) {
            int r = i >> 7, k = i & 127;
            dstT[r * SX_STRIDE + k] = (_Float16)srcT[i];
        }
    }
    __syncthreads();

    // ---- Load A fragments (16x128 of x, 4 K-steps of 32)
    Frag16 afr[4];
    {
        const _Float16* xrow = sX + wave * (16 * SX_STRIDE) + nl * SX_STRIDE;
#pragma unroll
        for (int ks = 0; ks < 4; ++ks) {
            int k0 = ks * 32 + hi * 8;
            afr[ks].h[0] = *(const v8h*)(xrow + k0);
            afr[ks].h[1] = *(const v8h*)(xrow + k0 + 16);
        }
    }

    // ---- Three gate GEMMs: 3 x 8 N-tiles x 4 K-steps = 96 WMMAs per wave
#pragma unroll
    for (int g = 0; g < 3; ++g) {
#pragma unroll
        for (int j = 0; j < 8; ++j) {
            const int col = j * 16 + nl;
            const _Float16* wrow = sWt + g * (C2 * WT_STRIDE) + col * WT_STRIDE;
            v8f acc = {};
#pragma unroll
            for (int ks = 0; ks < 4; ++ks) {
                Frag16 bfr;
                int k0 = ks * 32 + hi * 8;
                bfr.h[0] = *(const v8h*)(wrow + k0);
                bfr.h[1] = *(const v8h*)(wrow + k0 + 16);
                acc = __builtin_amdgcn_wmma_f32_16x16x32_f16(
                        false, afr[ks].v, false, bfr.v,
                        (short)0, acc, false, false);
            }
            // activation + stash to LDS (acc elem v -> row v+8*hi, col)
            const float bias = bg[g][col];
            _Float16* gdst = sG + (wave * 3 + g) * (16 * C2) + col;
#pragma unroll
            for (int v = 0; v < 8; ++v) {
                float val = acc[v] + bias;
                if (g == 0)      val = fast_sigmoid(val + 2.0f);    // forget
                else if (g == 1) val = fast_sigmoid(val);           // input
                else             val = fast_softplus(val) * 0.1f;   // delta
                gdst[(v + 8 * hi) * C2] = (_Float16)val;
            }
        }
    }
    // sG/sXf32 traffic below is wave-local; compiler-inserted s_wait_dscnt
    // orders same-wave LDS dependencies.

    // ---- Epilogue: complex input transform + unrolled T=16 scan
    const _Float16* gF = sG + (wave * 3 + 0) * (16 * C2);
    const _Float16* gI = sG + (wave * 3 + 1) * (16 * C2);
    const _Float16* gD = sG + (wave * 3 + 2) * (16 * C2);
    const float*    xT = sXf32 + wave * (16 * C2);

    for (int r = 0; r < 16; ++r) {
        const int rowg = row0 + r;
        const int hw   = rowg & (HW - 1);
        float*    op   = out + (size_t)rowg * C2;
        const int ab   = hw * Cdim;
#pragma unroll
        for (int ch = 0; ch < 2; ++ch) {
            const int c = ch * 32 + lane;   // 0..63, coalesced

            const float Ar = wsAr[ab + c];
            const float Ai = wsAi[ab + c];
            const float Br = wsBr[ab + c];
            const float Bi = wsBi[ab + c];

            const float xr = xT[r * C2 + c];
            const float xi = xT[r * C2 + Cdim + c];

            const float fr  = (float)gF[r * C2 + c];
            const float fi  = (float)gF[r * C2 + Cdim + c];
            const float igr = (float)gI[r * C2 + c];
            const float igi = (float)gI[r * C2 + Cdim + c];
            const float dlr = (float)gD[r * C2 + c];
            const float dli = (float)gD[r * C2 + Cdim + c];

            const float Bxr = Br * xr - Bi * xi;
            const float Bxi = Br * xi + Bi * xr;
            const float inr = dlr * igr * Bxr;
            const float ini = dli * igi * Bxi;

            float hr = 0.f, hcomp = 0.f;
#pragma unroll
            for (int t = 0; t < Tlen; ++t) {
                float Ahr = Ar * hr - Ai * hcomp;
                float Ahi = Ar * hcomp + Ai * hr;
                hr    = fr * Ahr + inr;
                hcomp = fi * Ahi + ini;
            }
            op[c]        = hr;
            op[Cdim + c] = hcomp;
        }
    }
}

// ---------------------------------------------------------------------------
extern "C" void kernel_launch(void* const* d_in, const int* in_sizes, int n_in,
                              void* d_out, int out_size, void* d_ws, size_t ws_size,
                              hipStream_t stream) {
    const float* x      = (const float*)d_in[0];
    const float* Wf     = (const float*)d_in[1];
    const float* bf     = (const float*)d_in[2];
    const float* Wi     = (const float*)d_in[3];
    const float* bi     = (const float*)d_in[4];
    const float* Wd     = (const float*)d_in[5];
    const float* bd     = (const float*)d_in[6];
    const float* A_real = (const float*)d_in[7];
    const float* A_imag = (const float*)d_in[8];
    const float* B_real = (const float*)d_in[9];
    const float* B_imag = (const float*)d_in[10];
    float* out = (float*)d_out;

    float* ws   = (float*)d_ws;
    float* wsAr = ws;
    float* wsAi = ws + (size_t)HW * Cdim;
    float* wsBr = ws + (size_t)2 * HW * Cdim;
    float* wsBi = ws + (size_t)3 * HW * Cdim;

    // Kernel 1: DFT tables (16384 blocks x 64 threads)
    dft_stab_kernel<<<dim3(HW), dim3(Cdim), 0, stream>>>(
        A_real, A_imag, B_real, B_imag, wsAr, wsAi, wsBr, wsBi);

    // Kernel 2: fused gates + scan. 262144 rows / (16 rows * 4 waves) = 4096 blocks.
    fused_gates_scan_kernel<<<dim3(NROWS / 64), dim3(128), SMEM_BYTES, stream>>>(
        x, Wf, bf, Wi, bi, Wd, bd, wsAr, wsAi, wsBr, wsBi, out);
}
